// Translator_45423574123025
// MI455X (gfx1250) — matrix operand order repaired and checked by reading, not compiled, not run on hardware
//
#include <hip/hip_runtime.h>
#include <math.h>

// ---------------------------------------------------------------------------
// CDNA5 (gfx1250) implementation of the transformer beam-search reference.
//
// Key algorithmic transform: the reference decoder consumes the FIXED label
// history `hi`; with the causal mask, position t of any length-L decode is
// identical for all L > t.  So all 27 beam steps (and dec0) are prefixes of a
// single length-28, batch-4 causal decode.  We run encoder once, decoder once,
// build a [28*4, 6] log-prob table, and replay the beam recurrence exactly.
//
// All GEMMs use v_wmma_f32_16x16x32_f16 (f16 operands, f32 accumulate).
// GEMM: 2x2 register-blocked (32x32 C per wave, 4 WMMAs per K-block),
// clamped (branch-free) M/N edges, float4-vectorized K blocks (batched
// global_load_b128 under s_clause), select-masked K tail.
// ---------------------------------------------------------------------------

typedef __attribute__((ext_vector_type(16))) _Float16 v16h;
typedef __attribute__((ext_vector_type(8)))  float    v8f;

#define DM       512
#define NHEAD    8
#define DH       64
#define DFF      2048
#define NVOCAB   6
#define NBEAM    4
#define MAXLEN   29
#define SLEN     900
#define BLANKTOK 5
#define LDEC     28              // positions 0..27 cover dec0 + steps 2..28
#define BROWS    (LDEC * NBEAM)  // 112, row = l*NBEAM + b
#define SQRT_D   22.627416997969522f
#define LN1E4    9.210340371976184f

// ---- fragment loaders -----------------------------------------------------
// A 16x32 f16 fragment: lane (hw,lr) holds row M=lr, K = hw*16 + {0..15} contiguous
__device__ __forceinline__ v16h load_a(const float* __restrict__ Arow,
                                       int k0, int hw)
{
    const float* pa = Arow + k0 + hw * 16;
    float4 a0 = *(const float4*)(pa);
    float4 a1 = *(const float4*)(pa + 4);
    float4 a2 = *(const float4*)(pa + 8);
    float4 a3 = *(const float4*)(pa + 12);
    v16h f;
    f[0]=(_Float16)a0.x; f[1]=(_Float16)a0.y; f[2]=(_Float16)a0.z; f[3]=(_Float16)a0.w;
    f[4]=(_Float16)a1.x; f[5]=(_Float16)a1.y; f[6]=(_Float16)a1.z; f[7]=(_Float16)a1.w;
    f[8]=(_Float16)a2.x; f[9]=(_Float16)a2.y; f[10]=(_Float16)a2.z; f[11]=(_Float16)a2.w;
    f[12]=(_Float16)a3.x; f[13]=(_Float16)a3.y; f[14]=(_Float16)a3.z; f[15]=(_Float16)a3.w;
    return f;
}

// B 32x16 f16 fragment (NT): lane holds col N, K = hw*16 + {0..15} contiguous
__device__ __forceinline__ v16h load_b_nt(const float* __restrict__ Brow,
                                          int k0, int hw)
{
    const float* pb = Brow + k0 + hw * 16;
    float4 b0 = *(const float4*)(pb);
    float4 b1 = *(const float4*)(pb + 4);
    float4 b2 = *(const float4*)(pb + 8);
    float4 b3 = *(const float4*)(pb + 12);
    v16h f;
    f[0]=(_Float16)b0.x; f[1]=(_Float16)b0.y; f[2]=(_Float16)b0.z; f[3]=(_Float16)b0.w;
    f[4]=(_Float16)b1.x; f[5]=(_Float16)b1.y; f[6]=(_Float16)b1.z; f[7]=(_Float16)b1.w;
    f[8]=(_Float16)b2.x; f[9]=(_Float16)b2.y; f[10]=(_Float16)b2.z; f[11]=(_Float16)b2.w;
    f[12]=(_Float16)b3.x; f[13]=(_Float16)b3.y; f[14]=(_Float16)b3.z; f[15]=(_Float16)b3.w;
    return f;
}

// B fragment (NN): strided rows of W; coalesced across lanes (ncol = lane).
__device__ __forceinline__ v16h load_b_nn(const float* __restrict__ W,
                                          long ldb, int ncol, int k0, int hw)
{
    const float* pb = W + (long)(k0 + hw * 16) * ldb + ncol;
    v16h f;
#pragma unroll
    for (int e = 0; e < 16; ++e) f[e] = (_Float16)pb[(long)e * ldb];
    return f;
}

// Tail loaders: clamped addresses + select-to-zero (branch-free).
__device__ __forceinline__ v16h load_a_tail(const float* __restrict__ Arow,
                                            int kfull, int K, int hw)
{
    v16h f;
#pragma unroll
    for (int e = 0; e < 16; ++e) {
        int kk = kfull + hw * 16 + e;
        int kc = kk < K ? kk : K - 1;
        float v = Arow[kc];
        f[e] = (_Float16)(kk < K ? v : 0.f);
    }
    return f;
}
__device__ __forceinline__ v16h load_b_tail(const float* __restrict__ W,
                                            long ldb, const float* __restrict__ Brow,
                                            int nt, int ncol, int kfull, int K, int hw)
{
    v16h f;
#pragma unroll
    for (int e = 0; e < 16; ++e) {
        int kk = kfull + hw * 16 + e;
        int kc = kk < K ? kk : K - 1;
        float v = nt ? Brow[kc] : W[(long)kc * ldb + ncol];
        f[e] = (_Float16)(kk < K ? v : 0.f);
    }
    return f;
}

// ---------------------------------------------------------------------------
// Generic WMMA GEMM:  C[m,n] = act( sum_k A[m,k] * B'[k,n] + bias[n] )
//   nt=1: B'[k,n] = W[n*ldb + k]   (linear layers, Q*K^T)
//   nt=0: B'[k,n] = W[k*ldb + n]   (P*V)
// z batching: zb = z/zdiv, zi = z%zdiv; each operand offset by zb*Out + zi*In.
// One wave computes a 32x32 C block (2x2 tiles, 4 WMMAs per 32-wide K block).
// Out-of-range rows/cols are CLAMPED for loads (their products only reach
// C entries that are never stored), so the hot loop is branch-free.
// All bases / leading dims / k-offsets are multiples of 4 floats -> float4 ok.
// ---------------------------------------------------------------------------
__global__ __launch_bounds__(32)
void gemm_wmma(const float* __restrict__ A, long lda,
               const float* __restrict__ W, long ldb,
               const float* __restrict__ bias,
               float* __restrict__ C, long ldc,
               int M, int N, int K, int nt, int relu, int zdiv,
               long aOut, long aIn, long bOut, long bIn, long cOut, long cIn)
{
    const int z  = blockIdx.z;
    const int zb = z / zdiv, zi = z % zdiv;
    A += (long)zb * aOut + (long)zi * aIn;
    W += (long)zb * bOut + (long)zi * bIn;
    C += (long)zb * cOut + (long)zi * cIn;

    const int tn   = blockIdx.x, tm = blockIdx.y;
    const int lane = threadIdx.x & 31;
    const int hw   = lane >> 4;          // half-wave: 0 or 1
    const int lr   = lane & 15;

    const int m0u = tm * 32 + lr,      m1u = m0u + 16;   // unclamped (stores)
    const int n0u = tn * 32 + lr,      n1u = n0u + 16;
    const int m0  = m0u < M ? m0u : M - 1;
    const int m1  = m1u < M ? m1u : M - 1;
    const int n0  = n0u < N ? n0u : N - 1;
    const int n1  = n1u < N ? n1u : N - 1;

    const float* __restrict__ Arow0 = A + (long)m0 * lda;
    const float* __restrict__ Arow1 = A + (long)m1 * lda;
    const float* __restrict__ Brow0 = W + (long)n0 * ldb;   // nt path bases
    const float* __restrict__ Brow1 = W + (long)n1 * ldb;

    v8f c00 = {}, c01 = {}, c10 = {}, c11 = {};
    const int kfull = K & ~31;

    for (int k0 = 0; k0 < kfull; k0 += 32) {
        v16h a0 = load_a(Arow0, k0, hw);
        v16h a1 = load_a(Arow1, k0, hw);
        v16h b0 = nt ? load_b_nt(Brow0, k0, hw) : load_b_nn(W, ldb, n0, k0, hw);
        v16h b1 = nt ? load_b_nt(Brow1, k0, hw) : load_b_nn(W, ldb, n1, k0, hw);

        if (k0 + 32 < kfull) {   // prefetch next K block (global_prefetch_b8)
            __builtin_prefetch(Arow0 + k0 + 32 + hw * 8, 0, 3);
            if (nt) __builtin_prefetch(Brow0 + k0 + 32 + hw * 16, 0, 3);
        }

        c00 = __builtin_amdgcn_wmma_f32_16x16x32_f16(false, a0, false, b0, (short)0, c00, false, false);
        c01 = __builtin_amdgcn_wmma_f32_16x16x32_f16(false, a0, false, b1, (short)0, c01, false, false);
        c10 = __builtin_amdgcn_wmma_f32_16x16x32_f16(false, a1, false, b0, (short)0, c10, false, false);
        c11 = __builtin_amdgcn_wmma_f32_16x16x32_f16(false, a1, false, b1, (short)0, c11, false, false);
    }

    if (kfull < K) {
        v16h a0 = load_a_tail(Arow0, kfull, K, hw);
        v16h a1 = load_a_tail(Arow1, kfull, K, hw);
        v16h b0 = load_b_tail(W, ldb, Brow0, nt, n0, kfull, K, hw);
        v16h b1 = load_b_tail(W, ldb, Brow1, nt, n1, kfull, K, hw);
        c00 = __builtin_amdgcn_wmma_f32_16x16x32_f16(false, a0, false, b0, (short)0, c00, false, false);
        c01 = __builtin_amdgcn_wmma_f32_16x16x32_f16(false, a0, false, b1, (short)0, c01, false, false);
        c10 = __builtin_amdgcn_wmma_f32_16x16x32_f16(false, a1, false, b0, (short)0, c10, false, false);
        c11 = __builtin_amdgcn_wmma_f32_16x16x32_f16(false, a1, false, b1, (short)0, c11, false, false);
    }

    // Store: C layout per tile: lane (hw,lr) vgpr j -> row j + hw*8, col lr.
#pragma unroll
    for (int nj = 0; nj < 2; ++nj) {
        int n = nj ? n1u : n0u;
        if (n >= N) continue;
        float bb = bias ? bias[n] : 0.f;
#pragma unroll
        for (int mi = 0; mi < 2; ++mi) {
            v8f acc = nj ? (mi ? c11 : c01) : (mi ? c10 : c00);
#pragma unroll
            for (int j = 0; j < 8; ++j) {
                int m = tm * 32 + mi * 16 + j + hw * 8;
                if (m < M) {
                    float r = acc[j] + bb;
                    if (relu) r = fmaxf(r, 0.f);
                    C[(long)m * ldc + n] = r;
                }
            }
        }
    }
}

// ---------------------------------------------------------------------------
// Row softmax with scale and optional causal mask (col > row masked).
// One 256-thread block per row; blockIdx.y selects the z-batch matrix.
// ---------------------------------------------------------------------------
__global__ __launch_bounds__(256)
void softmax_rows(float* __restrict__ S, int cols, long zstride,
                  float scale, int causal)
{
    const int row = blockIdx.x;
    float* p = S + (long)blockIdx.y * zstride + (long)row * cols;
    __shared__ float red[256];
    const int t = threadIdx.x;

    float mx = -3.0e38f;
    for (int c = t; c < cols; c += 256) {
        float v = p[c] * scale;
        if (causal && c > row) v = -3.0e38f;
        mx = fmaxf(mx, v);
    }
    red[t] = mx; __syncthreads();
    for (int s = 128; s > 0; s >>= 1) {
        if (t < s) red[t] = fmaxf(red[t], red[t + s]);
        __syncthreads();
    }
    mx = red[0]; __syncthreads();

    float sum = 0.f;
    for (int c = t; c < cols; c += 256) {
        float v = p[c] * scale;
        if (causal && c > row) v = -3.0e38f;
        v = __expf(v - mx);
        p[c] = v;
        sum += v;
    }
    red[t] = sum; __syncthreads();
    for (int s = 128; s > 0; s >>= 1) {
        if (t < s) red[t] += red[t + s];
        __syncthreads();
    }
    float inv = 1.f / red[0];
    for (int c = t; c < cols; c += 256) p[c] *= inv;
}

// ---------------------------------------------------------------------------
// x = LayerNorm(x + r) * g + b  (eps = 1e-5), one block per row.
// ---------------------------------------------------------------------------
__global__ __launch_bounds__(256)
void ln_add(float* __restrict__ x, const float* __restrict__ r,
            const float* __restrict__ g, const float* __restrict__ b, int cols)
{
    const long row = blockIdx.x;
    float* px = x + row * cols;
    const float* pr = r + row * cols;
    __shared__ float red[256];
    const int t = threadIdx.x;

    float s = 0.f;
    for (int c = t; c < cols; c += 256) s += px[c] + pr[c];
    red[t] = s; __syncthreads();
    for (int k = 128; k > 0; k >>= 1) { if (t < k) red[t] += red[t + k]; __syncthreads(); }
    float mu = red[0] / cols; __syncthreads();

    float vs = 0.f;
    for (int c = t; c < cols; c += 256) {
        float v = px[c] + pr[c] - mu; vs += v * v;
    }
    red[t] = vs; __syncthreads();
    for (int k = 128; k > 0; k >>= 1) { if (t < k) red[t] += red[t + k]; __syncthreads(); }
    float rstd = rsqrtf(red[0] / cols + 1e-5f);

    for (int c = t; c < cols; c += 256) {
        float v = (px[c] + pr[c] - mu) * rstd;
        px[c] = v * g[c] + b[c];
    }
}

// Sinusoidal positional encoding, computed analytically.
__device__ __forceinline__ float pe_val(int pos, int c)
{
    int j = c & ~1;
    float div = __expf(-(float)j * (LN1E4 / (float)DM));
    float a = (float)pos * div;
    return (c & 1) ? __cosf(a) : __sinf(a);
}

__global__ void enc_input(const float* __restrict__ src, float* __restrict__ x)
{
    int i = blockIdx.x * 256 + threadIdx.x;
    if (i >= SLEN * DM) return;
    int s = i / DM, c = i % DM;
    x[i] = src[i] * SQRT_D + pe_val(s, c);
}

__global__ void dec_input(const int* __restrict__ hi,
                          const float* __restrict__ embed,
                          float* __restrict__ x)
{
    int i = blockIdx.x * 256 + threadIdx.x;
    if (i >= BROWS * DM) return;
    int row = i / DM, c = i % DM;
    int l = row / NBEAM, b = row % NBEAM;
    int tok = hi[b * MAXLEN + l];
    x[i] = embed[tok * DM + c] * SQRT_D + pe_val(l, c);
}

__global__ void logsoftmax6(const float* __restrict__ logits,
                            float* __restrict__ lp)
{
    int row = blockIdx.x * blockDim.x + threadIdx.x;
    if (row >= BROWS) return;
    const float* p = logits + row * NVOCAB;
    float mx = p[0];
    for (int v = 1; v < NVOCAB; ++v) mx = fmaxf(mx, p[v]);
    float s = 0.f;
    for (int v = 0; v < NVOCAB; ++v) s += __expf(p[v] - mx);
    float lse = mx + __logf(s);
    for (int v = 0; v < NVOCAB; ++v) lp[row * NVOCAB + v] = p[v] - lse;
}

// ---------------------------------------------------------------------------
// Beam search replay on the precomputed log-prob table lp[(l*NBEAM+b)*6 + v].
// top_k tie-breaking matches jax.lax.top_k (lower index wins). Single thread.
// ---------------------------------------------------------------------------
__global__ void beam_kernel(const float* __restrict__ lp, float* __restrict__ out)
{
    if (threadIdx.x != 0 || blockIdx.x != 0) return;

    float scores[NBEAM];
    int gen[NBEAM][MAXLEN];
    for (int b = 0; b < NBEAM; ++b)
        for (int t = 0; t < MAXLEN; ++t) gen[b][t] = BLANKTOK;

    // dec0: position 0, beam-batch 0  ->  top-4 of 6
    {
        const float* p = lp + (0 * NBEAM + 0) * NVOCAB;
        bool used[NVOCAB] = {false};
        for (int j = 0; j < NBEAM; ++j) {
            int bi = -1; float bv = 0.f; bool first = true;
            for (int v = 0; v < NVOCAB; ++v)
                if (!used[v] && (first || p[v] > bv)) { bv = p[v]; bi = v; first = false; }
            used[bi] = true;
            scores[j] = bv;
            gen[j][1] = bi;
        }
    }

    for (int step = 2; step < MAXLEN; ++step) {
        float k2v[NBEAM][NBEAM]; int k2i[NBEAM][NBEAM];
        for (int b = 0; b < NBEAM; ++b) {
            const float* p = lp + ((step - 1) * NBEAM + b) * NVOCAB;
            bool used[NVOCAB] = {false};
            for (int j = 0; j < NBEAM; ++j) {
                int bi = -1; float bv = 0.f; bool first = true;
                for (int v = 0; v < NVOCAB; ++v)
                    if (!used[v] && (first || p[v] > bv)) { bv = p[v]; bi = v; first = false; }
                used[bi] = true;
                k2v[b][j] = bv; k2i[b][j] = bi;
            }
        }
        float cand[NBEAM * NBEAM];
        for (int b = 0; b < NBEAM; ++b)
            for (int j = 0; j < NBEAM; ++j)
                cand[b * NBEAM + j] = k2v[b][j] + scores[b];

        float ns[NBEAM]; int nf[NBEAM];
        bool used[NBEAM * NBEAM] = {false};
        for (int j = 0; j < NBEAM; ++j) {
            int bi = -1; float bv = 0.f; bool first = true;
            for (int q = 0; q < NBEAM * NBEAM; ++q)
                if (!used[q] && (first || cand[q] > bv)) { bv = cand[q]; bi = q; first = false; }
            used[bi] = true;
            ns[j] = bv; nf[j] = bi;
        }
        int ng[NBEAM][MAXLEN];
        for (int j = 0; j < NBEAM; ++j) {
            int r = nf[j] / NBEAM, c = nf[j] % NBEAM;
            for (int t = 0; t < MAXLEN; ++t) ng[j][t] = gen[r][t];
            ng[j][step] = k2i[r][c];
        }
        for (int j = 0; j < NBEAM; ++j) {
            scores[j] = ns[j];
            for (int t = 0; t < MAXLEN; ++t) gen[j][t] = ng[j][t];
        }
    }

    for (int t = 0; t < MAXLEN; ++t) out[t] = (float)gen[0][t];
    for (int j = 0; j < NBEAM; ++j) out[MAXLEN + j] = scores[j];
}

// ---------------------------------------------------------------------------
// Host side
// ---------------------------------------------------------------------------
static inline void gemm(hipStream_t s, const float* A, long lda,
                        const float* W, long ldb, const float* bias,
                        float* C, long ldc, int M, int N, int K,
                        int nt, int relu, int Z, int zdiv,
                        long aOut, long aIn, long bOut, long bIn,
                        long cOut, long cIn)
{
    dim3 g((N + 31) / 32, (M + 31) / 32, Z);
    gemm_wmma<<<g, 32, 0, s>>>(A, lda, W, ldb, bias, C, ldc, M, N, K,
                               nt, relu, zdiv, aOut, aIn, bOut, bIn, cOut, cIn);
}

// Workspace layout (float offsets).  Encoder score buffer is reused for all
// decoder scratch once the encoder is done.  Total ~44.4 MB.
enum : long {
    O_X    = 0,                  // 900*512   encoder activations / enc_out
    O_QKV  = 460800,             // 900*1536
    O_HID  = 1843200,            // 900*2048
    O_TMP  = 3686400,            // 900*512
    O_SC   = 4147200,            // 8*900*900 = 6480000
    O_ATT  = 10627200,           // 900*512
    // decoder region aliased inside O_SC:
    O_XD   = O_SC + 0,           // 112*512
    O_QKVD = O_SC + 57344,       // 112*1536
    O_SCD  = O_SC + 229376,      // 32*28*28
    O_ATTD = O_SC + 254464,      // 112*512
    O_TMPD = O_SC + 311808,      // 112*512
    O_QD   = O_SC + 369152,      // 112*512
    O_HIDD = O_SC + 426496,      // 112*2048
    O_KV   = O_SC + 655872,      // 900*1024
    O_CSC  = O_SC + 1577472,     // 32*28*900
    O_LOG  = O_SC + 2383872,     // 112*6
    O_LP   = O_SC + 2384544,     // 112*6
};

extern "C" void kernel_launch(void* const* d_in, const int* in_sizes, int n_in,
                              void* d_out, int out_size, void* d_ws, size_t ws_size,
                              hipStream_t stream)
{
    // Input order: top-level setup dict in insertion order, `params` leaves in
    // jax pytree order (dict keys sorted, lists in index order):
    //   0 src_seq, 1 hi(int), 2 trg_mask(bool, unused; causal mask is inlined)
    //   dec layer i (base 3+i*20): cproj_b, cproj_w, ff1_b, ff1_w, ff2_b,
    //     ff2_w, kv_b, kv_w, ln1_b, ln1_g, ln2_b, ln2_g, ln3_b, ln3_g,
    //     proj_b, proj_w, q_b, q_w, qkv_b, qkv_w
    //   43 embed
    //   enc layer i (base 44+i*12): ff1_b, ff1_w, ff2_b, ff2_w, ln1_b, ln1_g,
    //     ln2_b, ln2_g, proj_b, proj_w, qkv_b, qkv_w
    //   92 out_b, 93 out_w
    auto F = [&](int i) { return (const float*)d_in[i]; };
    const float* src   = F(0);
    const int*   hi    = (const int*)d_in[1];
    const float* embed = F(43);

    float* ws  = (float*)d_ws;
    float* X   = ws + O_X;   float* QKV  = ws + O_QKV;  float* HID  = ws + O_HID;
    float* TMP = ws + O_TMP; float* SC   = ws + O_SC;   float* ATT  = ws + O_ATT;
    float* XD  = ws + O_XD;  float* QKVD = ws + O_QKVD; float* SCD  = ws + O_SCD;
    float* ATTD= ws + O_ATTD;float* TMPD = ws + O_TMPD; float* QD   = ws + O_QD;
    float* HIDD= ws + O_HIDD;float* KV   = ws + O_KV;   float* CSC  = ws + O_CSC;
    float* LOG = ws + O_LOG; float* LP   = ws + O_LP;

    const float ATTN_SCALE = 0.125f;   // 1/sqrt(64)

    // ---------------- Encoder: S=900, B=1 ----------------
    enc_input<<<(SLEN * DM + 255) / 256, 256, 0, stream>>>(src, X);
    for (int l = 0; l < 4; ++l) {
        const int eb = 44 + l * 12;
        const float *ff1_b=F(eb+0), *ff1_w=F(eb+1), *ff2_b=F(eb+2), *ff2_w=F(eb+3);
        const float *ln1_b=F(eb+4), *ln1_g=F(eb+5), *ln2_b=F(eb+6), *ln2_g=F(eb+7);
        const float *proj_b=F(eb+8), *proj_w=F(eb+9), *qkv_b=F(eb+10), *qkv_w=F(eb+11);

        // qkv = x @ qkv_w^T + qkv_b                     [900,1536]
        gemm(stream, X, DM, qkv_w, DM, qkv_b, QKV, 1536, SLEN, 1536, DM,
             1, 0, 1, 1, 0,0, 0,0, 0,0);
        // per-head scores = Qh @ Kh^T                   8 x [900,900]
        gemm(stream, QKV, 1536, QKV + DM, 1536, nullptr, SC, SLEN,
             SLEN, SLEN, DH, 1, 0, NHEAD, NHEAD, 0, DH, 0, DH, 0, (long)SLEN*SLEN);
        softmax_rows<<<dim3(SLEN, NHEAD), 256, 0, stream>>>(
            SC, SLEN, (long)SLEN * SLEN, ATTN_SCALE, 0);
        // attn = P @ Vh                                 [900,512]
        gemm(stream, SC, SLEN, QKV + 2 * DM, 1536, nullptr, ATT, DM,
             SLEN, DH, SLEN, 0, 0, NHEAD, NHEAD, 0, (long)SLEN*SLEN, 0, DH, 0, DH);
        // proj + residual LN
        gemm(stream, ATT, DM, proj_w, DM, proj_b, TMP, DM, SLEN, DM, DM,
             1, 0, 1, 1, 0,0, 0,0, 0,0);
        ln_add<<<SLEN, 256, 0, stream>>>(X, TMP, ln1_g, ln1_b, DM);
        // FFN
        gemm(stream, X, DM, ff1_w, DM, ff1_b, HID, DFF, SLEN, DFF, DM,
             1, 1, 1, 1, 0,0, 0,0, 0,0);
        gemm(stream, HID, DFF, ff2_w, DFF, ff2_b, TMP, DM, SLEN, DM, DFF,
             1, 0, 1, 1, 0,0, 0,0, 0,0);
        ln_add<<<SLEN, 256, 0, stream>>>(X, TMP, ln2_g, ln2_b, DM);
    }

    // ---------------- Decoder: one causal pass, L=28, B=4 ----------------
    // Row layout matches [L, B, d] flattening: row = l*NBEAM + b.
    dec_input<<<(BROWS * DM + 255) / 256, 256, 0, stream>>>(hi, embed, XD);
    for (int l = 0; l < 2; ++l) {
        const int db = 3 + l * 20;
        const float *cproj_b=F(db+0), *cproj_w=F(db+1);
        const float *ff1_b=F(db+2), *ff1_w=F(db+3), *ff2_b=F(db+4), *ff2_w=F(db+5);
        const float *kv_b=F(db+6), *kv_w=F(db+7);
        const float *ln1_b=F(db+8), *ln1_g=F(db+9), *ln2_b=F(db+10), *ln2_g=F(db+11);
        const float *ln3_b=F(db+12), *ln3_g=F(db+13);
        const float *proj_b=F(db+14), *proj_w=F(db+15);
        const float *q_b=F(db+16), *q_w=F(db+17);
        const float *qkv_b=F(db+18), *qkv_w=F(db+19);

        // --- masked self-attention ---
        gemm(stream, XD, DM, qkv_w, DM, qkv_b, QKVD, 1536, BROWS, 1536, DM,
             1, 0, 1, 1, 0,0, 0,0, 0,0);
        // z = b*8+h; Q[l,k] = QKVD[(l*4+b)*1536 + h*64 + k]
        gemm(stream, QKVD, 4 * 1536, QKVD + DM, 4 * 1536, nullptr, SCD, LDEC,
             LDEC, LDEC, DH, 1, 0, NBEAM * NHEAD, NHEAD,
             1536, DH, 1536, DH, (long)NHEAD * LDEC * LDEC, (long)LDEC * LDEC);
        softmax_rows<<<dim3(LDEC, NBEAM * NHEAD), 256, 0, stream>>>(
            SCD, LDEC, (long)LDEC * LDEC, ATTN_SCALE, /*causal=*/1);
        gemm(stream, SCD, LDEC, QKVD + 2 * DM, 4 * 1536, nullptr, ATTD, 4 * DM,
             LDEC, DH, LDEC, 0, 0, NBEAM * NHEAD, NHEAD,
             (long)NHEAD * LDEC * LDEC, (long)LDEC * LDEC, 1536, DH, DM, DH);
        gemm(stream, ATTD, DM, proj_w, DM, proj_b, TMPD, DM, BROWS, DM, DM,
             1, 0, 1, 1, 0,0, 0,0, 0,0);
        ln_add<<<BROWS, 256, 0, stream>>>(XD, TMPD, ln1_g, ln1_b, DM);

        // --- cross-attention (enc_rep == enc for every beam) ---
        gemm(stream, X, DM, kv_w, DM, kv_b, KV, 1024, SLEN, 1024, DM,
             1, 0, 1, 1, 0,0, 0,0, 0,0);
        gemm(stream, XD, DM, q_w, DM, q_b, QD, DM, BROWS, DM, DM,
             1, 0, 1, 1, 0,0, 0,0, 0,0);
        gemm(stream, QD, 4 * DM, KV, 1024, nullptr, CSC, SLEN,
             LDEC, SLEN, DH, 1, 0, NBEAM * NHEAD, NHEAD,
             DM, DH, 0, DH, (long)NHEAD * LDEC * SLEN, (long)LDEC * SLEN);
        softmax_rows<<<dim3(LDEC, NBEAM * NHEAD), 256, 0, stream>>>(
            CSC, SLEN, (long)LDEC * SLEN, ATTN_SCALE, 0);
        gemm(stream, CSC, SLEN, KV + DM, 1024, nullptr, ATTD, 4 * DM,
             LDEC, DH, SLEN, 0, 0, NBEAM * NHEAD, NHEAD,
             (long)NHEAD * LDEC * SLEN, (long)LDEC * SLEN, 0, DH, DM, DH);
        gemm(stream, ATTD, DM, cproj_w, DM, cproj_b, TMPD, DM, BROWS, DM, DM,
             1, 0, 1, 1, 0,0, 0,0, 0,0);
        ln_add<<<BROWS, 256, 0, stream>>>(XD, TMPD, ln2_g, ln2_b, DM);

        // --- FFN ---
        gemm(stream, XD, DM, ff1_w, DM, ff1_b, HIDD, DFF, BROWS, DFF, DM,
             1, 1, 1, 1, 0,0, 0,0, 0,0);
        gemm(stream, HIDD, DFF, ff2_w, DFF, ff2_b, TMPD, DM, BROWS, DM, DFF,
             1, 0, 1, 1, 0,0, 0,0, 0,0);
        ln_add<<<BROWS, 256, 0, stream>>>(XD, TMPD, ln3_g, ln3_b, DM);
    }

    // Output projection + log-softmax -> lp table [28*4, 6]
    gemm(stream, XD, DM, F(93), DM, F(92), LOG, NVOCAB, BROWS, NVOCAB, DM,
         1, 0, 1, 1, 0,0, 0,0, 0,0);
    logsoftmax6<<<4, 32, 0, stream>>>(LOG, LP);

    // Beam-search replay -> d_out[0..28] = gen_seq[0], d_out[29..32] = scores
    beam_kernel<<<1, 1, 0, stream>>>(LP, (float*)d_out);

    (void)in_sizes; (void)n_in; (void)out_size; (void)ws_size;
}
